// DeepSeekMini_85504208929569
// MI455X (gfx1250) — compile-verified, matrix operand order checked by reading, not run on hardware
//
#include <hip/hip_runtime.h>
#include <hip/hip_bf16.h>
#include <stdint.h>

// Problem dims (fixed by the reference)
#define B_ 2
#define S_ 2048
#define H_ 2048
#define F_ 4096
#define E_ 8
#define T_ (B_ * S_)   // 4096 tokens
#define EPS_ 1e-6f
#define BAL_FACTOR_ 1e-4f

typedef __attribute__((ext_vector_type(16))) __bf16 v16bf;
typedef __attribute__((ext_vector_type(8)))  float  v8f;

union Q16 { uint4 q; __bf16 h[8]; };

// ---------------------------------------------------------------------------
// CDNA5 async copy: global -> LDS without touching VGPR data, ASYNCcnt-tracked.
// Generic pointers to __shared__ carry the LDS byte offset in the low 32 bits.
// ---------------------------------------------------------------------------
__device__ __forceinline__ void async_copy_b128(const void* gaddr, void* lds) {
  uint32_t l = (uint32_t)(uintptr_t)lds;
  asm volatile("global_load_async_to_lds_b128 %0, %1, off"
               :: "v"(l), "v"(gaddr) : "memory");
}
__device__ __forceinline__ void wait_async0() {
  asm volatile("s_wait_asynccnt 0" ::: "memory");
}

// Fragment read (LDS or global): 16-bit 16x32 WMMA operand layout (ISA 7.12.2):
// lanes 0-15 hold row M=lane%16, K={koff..koff+7, koff+16..koff+23}, koff=8*(lane/16).
__device__ __forceinline__ v16bf load_frag_bf16(const __bf16* p) {
  Q16 lo, hi;
  lo.q = *(const uint4*)(p);
  hi.q = *(const uint4*)(p + 16);
  v16bf a;
#pragma unroll
  for (int i = 0; i < 8; ++i) { a[i] = lo.h[i]; a[i + 8] = hi.h[i]; }
  return a;
}

__device__ __forceinline__ v8f wmma_bf16(v16bf a, v16bf b, v8f c) {
  return __builtin_amdgcn_wmma_f32_16x16x32_bf16(
      false, a, false, b, (short)0, c, false, false);
}

__device__ __forceinline__ float wave_sum(float v) {
#pragma unroll
  for (int off = 16; off > 0; off >>= 1) v += __shfl_xor(v, off, 32);
  return v;
}

__device__ __forceinline__ float fast_silu(float x) {
  return x * __builtin_amdgcn_rcpf(1.0f + __expf(-x));   // v_rcp_f32, no div chain
}

// ---------------------------------------------------------------------------
// Kernel 0: zero the router load accumulator
// ---------------------------------------------------------------------------
__global__ void ws_init_kernel(float* load_acc) {
  if (threadIdx.x < E_) load_acc[threadIdx.x] = 0.0f;
}

// ---------------------------------------------------------------------------
// Kernel 1: fp32 -> bf16 weight conversion (bandwidth pass, 8 elems/thread)
// ---------------------------------------------------------------------------
__global__ __launch_bounds__(256)
void cvt_bf16_kernel(const float* __restrict__ src, __bf16* __restrict__ dst) {
  size_t i = ((size_t)blockIdx.x * 256 + threadIdx.x) * 8;
  float4 a = *(const float4*)(src + i);
  float4 b = *(const float4*)(src + i + 4);
  Q16 q;
  q.h[0] = (__bf16)a.x; q.h[1] = (__bf16)a.y; q.h[2] = (__bf16)a.z; q.h[3] = (__bf16)a.w;
  q.h[4] = (__bf16)b.x; q.h[5] = (__bf16)b.y; q.h[6] = (__bf16)b.z; q.h[7] = (__bf16)b.w;
  *(uint4*)(dst + i) = q.q;
}

// ---------------------------------------------------------------------------
// Kernel 2: RMSNorm + router. One wave per token (wave32).
// ---------------------------------------------------------------------------
__global__ __launch_bounds__(256)
void rms_router_kernel(const float* __restrict__ hs,
                       const float* __restrict__ norm_w,
                       const float* __restrict__ router_w,
                       const float* __restrict__ ebias,
                       __bf16* __restrict__ xb,
                       float* __restrict__ dispatch,
                       float* __restrict__ load_acc) {
  const int lane = threadIdx.x & 31;
  const int wave = threadIdx.x >> 5;
  const int t    = blockIdx.x * 8 + wave;
  const float* x = hs + (size_t)t * H_;

  float ss = 0.0f;
  for (int j = lane; j < H_; j += 32) { float v = x[j]; ss += v * v; }
  ss = wave_sum(ss);
  const float rinv = rsqrtf(ss * (1.0f / H_) + EPS_);

  float acc[E_];
#pragma unroll
  for (int e = 0; e < E_; ++e) acc[e] = 0.0f;
  for (int j = lane; j < H_; j += 32) {
    float v = x[j] * rinv * norm_w[j];
    xb[(size_t)t * H_ + j] = (__bf16)v;
#pragma unroll
    for (int e = 0; e < E_; ++e) acc[e] += v * router_w[(size_t)e * H_ + j];
  }
  float logit[E_];
#pragma unroll
  for (int e = 0; e < E_; ++e) logit[e] = wave_sum(acc[e]) + ebias[e];

  float mx = logit[0];
#pragma unroll
  for (int e = 1; e < E_; ++e) mx = fmaxf(mx, logit[e]);
  float p[E_]; float psum = 0.0f;
#pragma unroll
  for (int e = 0; e < E_; ++e) { p[e] = __expf(logit[e] - mx); psum += p[e]; }
  const float inv = 1.0f / psum;
#pragma unroll
  for (int e = 0; e < E_; ++e) p[e] *= inv;

  int i1 = 0; float v1 = p[0];
#pragma unroll
  for (int e = 1; e < E_; ++e) { if (p[e] > v1) { v1 = p[e]; i1 = e; } }
  int i2 = -1; float v2 = -1.0f;
#pragma unroll
  for (int e = 0; e < E_; ++e) { if (e != i1 && p[e] > v2) { v2 = p[e]; i2 = e; } }
  const float rs = 1.0f / (v1 + v2);

  if (lane == 0) {
#pragma unroll
    for (int e = 0; e < E_; ++e) {
      float w = (e == i1) ? v1 * rs : ((e == i2) ? v2 * rs : 0.0f);
      dispatch[(size_t)t * E_ + e] = w;
      atomicAdd(&load_acc[e], p[e]);
    }
  }
}

// ---------------------------------------------------------------------------
// Kernel 3: up-projection + SwiGLU,  act = silu(x@w1^T) * (x@w3^T)   (K = H)
// Block tile BM=64 x BN=128 (per matrix); 8 waves as 2(M) x 4(N); wave tile
// 32x32 per matrix -> 8 WMMA/K-step. Async double-buffered LDS panels.
// All fragment DS loads hoisted ahead of the WMMA block so the scheduler can
// issue 12x ds_load_b128 back-to-back and retire WMMAs on graduated dscnt.
// LDS: A 64x32 bf16 (4KB) | B1 128x32 (8KB) | B3 128x32 (8KB) = 20KB x2
// ---------------------------------------------------------------------------
__global__ __launch_bounds__(256)
void gemm_act_kernel(const __bf16* __restrict__ xb,
                     const __bf16* __restrict__ w1,
                     const __bf16* __restrict__ w3,
                     __bf16* __restrict__ act) {
  __shared__ char sm[2][20480];
  const int tid  = threadIdx.x;
  const int tn0  = blockIdx.x * 128;
  const int tm0  = blockIdx.y * 64;

  // staging: 1 A chunk + 2 B1 + 2 B3 chunks per thread, 16B each
  const int rA = tid >> 2, cA = tid & 3;
  const char* gA = (const char*)(xb + (size_t)(tm0 + rA) * H_) + cA * 16;
  const int lA = rA * 64 + cA * 16;

  const int rB0 = (tid >> 2), rB1 = (tid >> 2) + 64, cB = tid & 3;
  const char* gB1_0 = (const char*)(w1 + (size_t)(tn0 + rB0) * H_) + cB * 16;
  const char* gB1_1 = (const char*)(w1 + (size_t)(tn0 + rB1) * H_) + cB * 16;
  const char* gB3_0 = (const char*)(w3 + (size_t)(tn0 + rB0) * H_) + cB * 16;
  const char* gB3_1 = (const char*)(w3 + (size_t)(tn0 + rB1) * H_) + cB * 16;
  const int lB1_0 = 4096  + rB0 * 64 + cB * 16;
  const int lB1_1 = 4096  + rB1 * 64 + cB * 16;
  const int lB3_0 = 12288 + rB0 * 64 + cB * 16;
  const int lB3_1 = 12288 + rB1 * 64 + cB * 16;

  // compute-fragment LDS offsets (bytes)
  const int lane = tid & 31, wave = tid >> 5;
  const int wm = wave >> 2, wn = wave & 3;
  const int frow = lane & 15;
  const int koffe = (lane >> 4) << 3;
  int aoff[2], b1off[2], b3off[2];
#pragma unroll
  for (int f = 0; f < 2; ++f) {
    aoff[f]  = ((wm * 32 + f * 16 + frow) * 32 + koffe) * 2;
    b1off[f] = 4096  + ((wn * 32 + f * 16 + frow) * 32 + koffe) * 2;
    b3off[f] = 12288 + ((wn * 32 + f * 16 + frow) * 32 + koffe) * 2;
  }

  v8f c1[2][2] = {}; v8f c3[2][2] = {};

  async_copy_b128(gA,    &sm[0][lA]);
  async_copy_b128(gB1_0, &sm[0][lB1_0]);
  async_copy_b128(gB1_1, &sm[0][lB1_1]);
  async_copy_b128(gB3_0, &sm[0][lB3_0]);
  async_copy_b128(gB3_1, &sm[0][lB3_1]);
  wait_async0();
  __syncthreads();

  int buf = 0;
  for (int k0 = 0; k0 < H_; k0 += 32) {
    if (k0 + 32 < H_) {
      const int kb = (k0 + 32) * 2;
      char* d = sm[buf ^ 1];
      async_copy_b128(gA + kb,    d + lA);
      async_copy_b128(gB1_0 + kb, d + lB1_0);
      async_copy_b128(gB1_1 + kb, d + lB1_1);
      async_copy_b128(gB3_0 + kb, d + lB3_0);
      async_copy_b128(gB3_1 + kb, d + lB3_1);
    }
    const char* s = sm[buf];
    // hoist ALL fragment loads (12x ds_load_b128 worth) before the WMMA block
    v16bf a0 = load_frag_bf16((const __bf16*)(s + aoff[0]));
    v16bf a1 = load_frag_bf16((const __bf16*)(s + aoff[1]));
    v16bf b10 = load_frag_bf16((const __bf16*)(s + b1off[0]));
    v16bf b11 = load_frag_bf16((const __bf16*)(s + b1off[1]));
    v16bf b30 = load_frag_bf16((const __bf16*)(s + b3off[0]));
    v16bf b31 = load_frag_bf16((const __bf16*)(s + b3off[1]));
    c1[0][0] = wmma_bf16(a0, b10, c1[0][0]);
    c1[1][0] = wmma_bf16(a1, b10, c1[1][0]);
    c1[0][1] = wmma_bf16(a0, b11, c1[0][1]);
    c1[1][1] = wmma_bf16(a1, b11, c1[1][1]);
    c3[0][0] = wmma_bf16(a0, b30, c3[0][0]);
    c3[1][0] = wmma_bf16(a1, b30, c3[1][0]);
    c3[0][1] = wmma_bf16(a0, b31, c3[0][1]);
    c3[1][1] = wmma_bf16(a1, b31, c3[1][1]);
    wait_async0();
    __syncthreads();
    buf ^= 1;
  }

  // epilogue: C element j of lane -> M = j + 8*(lane>=16), N = lane&15
  const int mb = ((lane >> 4) << 3);
#pragma unroll
  for (int fm = 0; fm < 2; ++fm)
#pragma unroll
    for (int fn = 0; fn < 2; ++fn)
#pragma unroll
      for (int j = 0; j < 8; ++j) {
        int row = tm0 + wm * 32 + fm * 16 + mb + j;
        int col = tn0 + wn * 32 + fn * 16 + frow;
        float h1 = c1[fm][fn][j], h3 = c3[fm][fn][j];
        act[(size_t)row * F_ + col] = (__bf16)(fast_silu(h1) * h3);
      }
}

// ---------------------------------------------------------------------------
// Kernel 4: down-projection (K = F). Block tile BM=64 x BN=256; 8 waves as
// 2(M) x 4(N); wave tile 32x64 -> 8 WMMA/K-step.
// eidx < 0: shared expert -> out = c + residual
// eidx >= 0: routed expert -> out += dispatch[t,e] * c
// LDS: A 64x32 (4KB) | B 256x32 (16KB) = 20KB x2
// ---------------------------------------------------------------------------
__global__ __launch_bounds__(256)
void gemm_down_kernel(const __bf16* __restrict__ act,
                      const __bf16* __restrict__ w2,
                      const float* __restrict__ residual,
                      const float* __restrict__ dispatch,
                      int eidx,
                      float* __restrict__ out) {
  __shared__ char sm[2][20480];
  const int tid = threadIdx.x;
  const int tn0 = blockIdx.x * 256;
  const int tm0 = blockIdx.y * 64;

  const int rA = tid >> 2, cc = tid & 3;
  const char* gA = (const char*)(act + (size_t)(tm0 + rA) * F_) + cc * 16;
  const int lA = rA * 64 + cc * 16;

  const char* gB[4]; int lB[4];
#pragma unroll
  for (int j = 0; j < 4; ++j) {
    int r = (tid >> 2) + j * 64;
    gB[j] = (const char*)(w2 + (size_t)(tn0 + r) * F_) + cc * 16;
    lB[j] = 4096 + r * 64 + cc * 16;
  }

  const int lane = tid & 31, wave = tid >> 5;
  const int wm = wave >> 2, wn = wave & 3;
  const int frow = lane & 15;
  const int koffe = (lane >> 4) << 3;
  int aoff[2], boff[4];
#pragma unroll
  for (int f = 0; f < 2; ++f)
    aoff[f] = ((wm * 32 + f * 16 + frow) * 32 + koffe) * 2;
#pragma unroll
  for (int f = 0; f < 4; ++f)
    boff[f] = 4096 + ((wn * 64 + f * 16 + frow) * 32 + koffe) * 2;

  v8f c[2][4] = {};

  async_copy_b128(gA, &sm[0][lA]);
#pragma unroll
  for (int j = 0; j < 4; ++j) async_copy_b128(gB[j], &sm[0][lB[j]]);
  wait_async0();
  __syncthreads();

  int buf = 0;
  for (int k0 = 0; k0 < F_; k0 += 32) {
    if (k0 + 32 < F_) {
      const int kb = (k0 + 32) * 2;
      char* d = sm[buf ^ 1];
      async_copy_b128(gA + kb, d + lA);
#pragma unroll
      for (int j = 0; j < 4; ++j) async_copy_b128(gB[j] + kb, d + lB[j]);
    }
    const char* s = sm[buf];
    // hoist all fragment loads before the WMMA block
    v16bf a0 = load_frag_bf16((const __bf16*)(s + aoff[0]));
    v16bf a1 = load_frag_bf16((const __bf16*)(s + aoff[1]));
    v16bf b0 = load_frag_bf16((const __bf16*)(s + boff[0]));
    v16bf b1 = load_frag_bf16((const __bf16*)(s + boff[1]));
    v16bf b2 = load_frag_bf16((const __bf16*)(s + boff[2]));
    v16bf b3 = load_frag_bf16((const __bf16*)(s + boff[3]));
    c[0][0] = wmma_bf16(a0, b0, c[0][0]);
    c[1][0] = wmma_bf16(a1, b0, c[1][0]);
    c[0][1] = wmma_bf16(a0, b1, c[0][1]);
    c[1][1] = wmma_bf16(a1, b1, c[1][1]);
    c[0][2] = wmma_bf16(a0, b2, c[0][2]);
    c[1][2] = wmma_bf16(a1, b2, c[1][2]);
    c[0][3] = wmma_bf16(a0, b3, c[0][3]);
    c[1][3] = wmma_bf16(a1, b3, c[1][3]);
    wait_async0();
    __syncthreads();
    buf ^= 1;
  }

  const int mb = ((lane >> 4) << 3);
#pragma unroll
  for (int fm = 0; fm < 2; ++fm)
#pragma unroll
    for (int j = 0; j < 8; ++j) {
      int row = tm0 + wm * 32 + fm * 16 + mb + j;
      float scale = (eidx >= 0) ? dispatch[(size_t)row * E_ + eidx] : 1.0f;
#pragma unroll
      for (int fn = 0; fn < 4; ++fn) {
        int col = tn0 + wn * 64 + fn * 16 + frow;
        size_t idx = (size_t)row * H_ + col;
        if (eidx < 0) out[idx] = c[fm][fn][j] + residual[idx];
        else          out[idx] += c[fm][fn][j] * scale;
      }
    }
}

// ---------------------------------------------------------------------------
// Kernel 5: balance loss scalar
// ---------------------------------------------------------------------------
__global__ void balance_kernel(const float* __restrict__ load_acc,
                               float* __restrict__ out_bal) {
  if (threadIdx.x == 0 && blockIdx.x == 0) {
    const float target = 1.0f / E_;
    float bal = 0.0f;
#pragma unroll
    for (int e = 0; e < E_; ++e) {
      float load = load_acc[e] * (1.0f / (float)T_);
      bal += target * (__logf(target) - __logf(load));
    }
    out_bal[0] = bal * (1.0f / E_) * BAL_FACTOR_;
  }
}

// ---------------------------------------------------------------------------
extern "C" void kernel_launch(void* const* d_in, const int* in_sizes, int n_in,
                              void* d_out, int out_size, void* d_ws, size_t ws_size,
                              hipStream_t stream) {
  const float* hs     = (const float*)d_in[0];
  const float* norm_w = (const float*)d_in[1];
  const float* rw     = (const float*)d_in[2];
  const float* ebias  = (const float*)d_in[3];
  const float* sw1    = (const float*)d_in[4];
  const float* sw2    = (const float*)d_in[5];
  const float* sw3    = (const float*)d_in[6];
  const float* ew1    = (const float*)d_in[7];
  const float* ew2    = (const float*)d_in[8];
  const float* ew3    = (const float*)d_in[9];
  float* out = (float*)d_out;

  // ws: xb[T,H]bf16 | act[T,F]bf16 | wb1[F,H]bf16 | wb3[F,H]bf16 | wb2[H,F]bf16
  //     | dispatch[T,E]f32 | load_acc[E]f32   (~96 MB total)
  char* ws = (char*)d_ws;
  __bf16* xb  = (__bf16*)ws;
  __bf16* act = (__bf16*)(ws + (size_t)T_ * H_ * 2);
  __bf16* wb1 = (__bf16*)(ws + (size_t)T_ * H_ * 2 + (size_t)T_ * F_ * 2);
  __bf16* wb3 = wb1 + (size_t)F_ * H_;
  __bf16* wb2 = wb3 + (size_t)F_ * H_;
  float* dispatch = (float*)(wb2 + (size_t)F_ * H_);
  float* load_acc = dispatch + (size_t)T_ * E_;

  const size_t WN = (size_t)F_ * H_;
  const int grid_cvt = (int)(WN / 2048);
  const dim3 grid_up(F_ / 128, T_ / 64);
  const dim3 grid_dn(H_ / 256, T_ / 64);

  ws_init_kernel<<<1, 32, 0, stream>>>(load_acc);
  rms_router_kernel<<<T_ / 8, 256, 0, stream>>>(hs, norm_w, rw, ebias,
                                                xb, dispatch, load_acc);

  for (int e = -1; e < E_; ++e) {
    const float* w1 = (e < 0) ? sw1 : ew1 + (size_t)e * WN;
    const float* w3 = (e < 0) ? sw3 : ew3 + (size_t)e * WN;
    const float* w2 = (e < 0) ? sw2 : ew2 + (size_t)e * WN;
    cvt_bf16_kernel<<<grid_cvt, 256, 0, stream>>>(w1, wb1);
    cvt_bf16_kernel<<<grid_cvt, 256, 0, stream>>>(w3, wb3);
    cvt_bf16_kernel<<<grid_cvt, 256, 0, stream>>>(w2, wb2);
    gemm_act_kernel<<<grid_up, 256, 0, stream>>>(xb, wb1, wb3, act);
    gemm_down_kernel<<<grid_dn, 256, 0, stream>>>(act, wb2, hs, dispatch, e, out);
  }

  balance_kernel<<<1, 1, 0, stream>>>(load_acc, out + (size_t)T_ * H_);
}
